// SDPA_10634339025524
// MI455X (gfx1250) — matrix operand order repaired and checked by reading, not compiled, not run on hardware
//
#include <hip/hip_runtime.h>

#define SEQ   2048
#define DH    64
#define NHEAD 16
#define NBAT  2
#define BH    (NBAT * NHEAD)
#define QTILE 16
#define QPW   32                 // queries per wave (2 subtiles)
#define KBLK  32
#define WAVES 8
#define QBLK  (QPW * WAVES)      // 256 queries per workgroup

typedef __attribute__((ext_vector_type(16))) __bf16 v16bf;
typedef __attribute__((ext_vector_type(8)))  float  v8f;
typedef int v4i __attribute__((vector_size(16)));   // matches async builtin param type

union BF16x16 { unsigned short u[16]; uint4 q[2]; v16bf v; };
union BF16x4  { unsigned short u[4];  uint2 q; };

#define AS1 __attribute__((address_space(1)))
#define AS3 __attribute__((address_space(3)))

#if defined(__gfx1250__) && __has_builtin(__builtin_amdgcn_global_load_async_to_lds_b128)
#define HAVE_ASYNC 1
#else
#define HAVE_ASYNC 0
#endif

__device__ __forceinline__ AS3 v4i* lds_cast(void* p) {
    return (AS3 v4i*)(unsigned int)(unsigned long long)p;
}
__device__ __forceinline__ AS1 v4i* glb_cast(const void* p) {
    return (AS1 v4i*)(unsigned long long)p;
}

#if HAVE_ASYNC
#if __has_builtin(__builtin_amdgcn_s_wait_asynccnt)
#define WAIT_ASYNC() __builtin_amdgcn_s_wait_asynccnt(0)
#else
#define WAIT_ASYNC() asm volatile("s_wait_asynccnt 0" ::: "memory")
#endif
#endif

__device__ __forceinline__ unsigned short f2bf(float f) {
    unsigned int u = __float_as_uint(f);
    unsigned int r = 0x7FFFu + ((u >> 16) & 1u);   // round-to-nearest-even
    return (unsigned short)((u + r) >> 16);
}

// ---------------------------------------------------------------------------
// Pre-pass: K -> bf16 transposed [bh][d][s], V -> bf16 [bh][s][d]
// ---------------------------------------------------------------------------
__global__ __launch_bounds__(256)
void cvt_bf16_kernel(const float* __restrict__ K, const float* __restrict__ V,
                     unsigned short* __restrict__ Kt, unsigned short* __restrict__ Vb) {
    const size_t i0 = ((size_t)blockIdx.x * 256 + threadIdx.x) * 4;
    {   // V: straight convert, vectorized
        const float4 v4 = *(const float4*)(V + i0);
        BF16x4 o;
        o.u[0] = f2bf(v4.x); o.u[1] = f2bf(v4.y); o.u[2] = f2bf(v4.z); o.u[3] = f2bf(v4.w);
        *(uint2*)(Vb + i0) = o.q;
    }
    {   // K: convert + per-(b,h) transpose [s][d] -> [d][s]
        const float4 k4 = *(const float4*)(K + i0);
        const size_t bh = i0 / ((size_t)SEQ * DH);
        const size_t r  = i0 - bh * (size_t)SEQ * DH;
        const int s = (int)(r / DH);
        const int d = (int)(r % DH);          // multiple of 4
        unsigned short* dst = Kt + bh * (size_t)DH * SEQ + (size_t)d * SEQ + s;
        dst[0 * SEQ] = f2bf(k4.x);
        dst[1 * SEQ] = f2bf(k4.y);
        dst[2 * SEQ] = f2bf(k4.z);
        dst[3 * SEQ] = f2bf(k4.w);
    }
}

// ---------------------------------------------------------------------------
// Flash attention: 8 waves x 32-query tiles (2 subtiles), 32-key blocks
// ---------------------------------------------------------------------------
__global__ __launch_bounds__(256)
void sdpa_fa_wmma_bf16(const float* __restrict__ Q,
                       const unsigned short* __restrict__ Ktg,  // [bh][d][s] bf16
                       const unsigned short* __restrict__ Vbg,  // [bh][s][d] bf16
                       float* __restrict__ O) {
    __shared__ unsigned short KtS[2][DH * KBLK];   // 2 x 4 KB, K^T tiles (d-major)
    __shared__ unsigned short VS [2][KBLK * DH];   // 2 x 4 KB, V tiles
    __shared__ float          Ss [WAVES][QTILE * KBLK]; // 16 KB, per-wave logit transpose

    const int lane = threadIdx.x & 31;
    const int wave = threadIdx.x >> 5;
    const int half = lane >> 4;
    const int ln   = lane & 15;

    const int bh   = blockIdx.y;
    const int q0wg = blockIdx.x * QBLK;
    const int q0   = q0wg + wave * QPW;            // first of 32 queries for this wave
    const size_t baseF  = (size_t)bh * SEQ * DH;
    const size_t baseKt = (size_t)bh * DH * SEQ;
    const size_t baseV  = (size_t)bh * SEQ * DH;

    auto issue_tiles = [&](int kb, int buf) {
        const int t = threadIdx.x;
        {   // K^T: 64 d-rows x 32 keys; 16B per thread
            const int row = t >> 2, seg = (t & 3) * 8;
            const unsigned short* src = Ktg + baseKt + (size_t)row * SEQ + kb * KBLK + seg;
            unsigned short* dst = &KtS[buf][row * KBLK + seg];
#if HAVE_ASYNC
            __builtin_amdgcn_global_load_async_to_lds_b128(glb_cast(src), lds_cast(dst), 0, 0);
#else
            *(uint4*)dst = *(const uint4*)src;
#endif
        }
        {   // V: 32 key-rows x 64 d; 16B per thread
            const int row = t >> 3, seg = (t & 7) * 8;
            const unsigned short* src = Vbg + baseV + (size_t)(kb * KBLK + row) * DH + seg;
            unsigned short* dst = &VS[buf][row * DH + seg];
#if HAVE_ASYNC
            __builtin_amdgcn_global_load_async_to_lds_b128(glb_cast(src), lds_cast(dst), 0, 0);
#else
            *(uint4*)dst = *(const uint4*)src;
#endif
        }
    };

    // ---- Q subtiles (2 x 16x64) -> A-operand layout, pre-scaled by 1/sqrt(64)
    v16bf Qa[2][2];
#pragma unroll
    for (int qt = 0; qt < 2; ++qt) {
        const float* qrow = Q + baseF + (size_t)(q0 + 16 * qt + ln) * DH;
#pragma unroll
        for (int c = 0; c < 2; ++c) {
            BF16x16 tmp;
#pragma unroll
            for (int e = 0; e < 16; ++e) {
                const int d = 32 * c + 16 * (e >> 3) + 8 * half + (e & 7);
                tmp.u[e] = f2bf(qrow[d] * 0.125f);
            }
            Qa[qt][c] = tmp.v;
        }
    }

    v8f o[2][4];
#pragma unroll
    for (int qt = 0; qt < 2; ++qt)
#pragma unroll
        for (int g = 0; g < 4; ++g) o[qt][g] = (v8f){};
    float mrow[2] = {-1e30f, -1e30f};
    float lrow[2] = {0.0f, 0.0f};

    const int nblk = (q0wg + QBLK) / KBLK;

#if HAVE_ASYNC
    issue_tiles(0, 0);
#endif
    for (int kb = 0; kb < nblk; ++kb) {
#if HAVE_ASYNC
        const int cur = kb & 1;
        WAIT_ASYNC();
        __syncthreads();
        if (kb + 1 < nblk) issue_tiles(kb + 1, cur ^ 1);
        if (kb + 2 < nblk) {
            __builtin_prefetch(Ktg + baseKt + (threadIdx.x >> 2) * SEQ + (kb + 2) * KBLK, 0, 3);
            __builtin_prefetch(Vbg + baseV + (size_t)((kb + 2) * KBLK + (threadIdx.x >> 3)) * DH, 0, 3);
        }
#else
        const int cur = 0;
        __syncthreads();
        issue_tiles(kb, 0);
        if (kb + 1 < nblk)
            __builtin_prefetch(Vbg + baseV + (size_t)((kb + 1) * KBLK) * DH, 0, 3);
        __syncthreads();
#endif
        const int kbase = kb * KBLK;
        // wave-uniform causal skip: block entirely in the future of all 32 rows
        if (kbase > q0 + QPW - 1) continue;

        const unsigned short* kt = KtS[cur];
        const unsigned short* vs = VS[cur];

        // ---- K^T as B-operand (shared across both q-subtiles)
        v16bf Kb[2][2];
#pragma unroll
        for (int c = 0; c < 2; ++c) {
            BF16x16 t0, t1;
#pragma unroll
            for (int e = 0; e < 16; ++e) {
                const int d = 32 * c + 16 * (e >> 3) + 8 * half + (e & 7);
                t0.u[e] = kt[d * KBLK + ln];
                t1.u[e] = kt[d * KBLK + 16 + ln];
            }
            Kb[0][c] = t0.v;
            Kb[1][c] = t1.v;
        }

        v16bf Pa[2];
        bool act[2];
#pragma unroll
        for (int qt = 0; qt < 2; ++qt) {
            const int qb = q0 + 16 * qt;          // subtile base row
            act[qt] = (kbase <= qb + QTILE - 1);  // wave-uniform
            if (!act[qt]) continue;

            // ---- logits: two 16x16 f32 tiles (C/D layout)
            v8f st[2];
#pragma unroll
            for (int n = 0; n < 2; ++n) {
                v8f acc = {};
                acc = __builtin_amdgcn_wmma_f32_16x16x32_bf16(false, Qa[qt][0], false, Kb[n][0],
                                                              (short)0, acc, false, false);
                acc = __builtin_amdgcn_wmma_f32_16x16x32_bf16(false, Qa[qt][1], false, Kb[n][1],
                                                              (short)0, acc, false, false);
                st[n] = acc;
            }

            // ---- transpose logits through per-wave LDS: Ss[row][key]
            float* sw = Ss[wave];
#pragma unroll
            for (int r = 0; r < 8; ++r) {
                sw[(r + 8 * half) * KBLK + ln]      = st[0][r];
                sw[(r + 8 * half) * KBLK + 16 + ln] = st[1][r];
            }
            float se[16];
            {
                const float4* p0 = (const float4*)&sw[ln * KBLK + 8 * half];
                const float4* p1 = (const float4*)&sw[ln * KBLK + 16 + 8 * half];
                *(float4*)&se[0]  = p0[0];
                *(float4*)&se[4]  = p0[1];
                *(float4*)&se[8]  = p1[0];
                *(float4*)&se[12] = p1[1];
            }
            // ---- causal mask only on diagonal blocks (wave-uniform branch)
            if (kbase + KBLK - 1 > qb) {
                const int qrow = qb + ln;
#pragma unroll
                for (int e = 0; e < 16; ++e) {
                    const int k = kbase + 16 * (e >> 3) + 8 * half + (e & 7);
                    if (k > qrow) se[e] = -1e30f;
                }
            }
            float t = se[0];
#pragma unroll
            for (int e = 1; e < 16; ++e) t = fmaxf(t, se[e]);
            t = fmaxf(t, __shfl_xor(t, 16, 32));
            const float mnew = fmaxf(mrow[qt], t);

            // ---- exp + rowsum + pack P directly in A-operand layout
            float rs = 0.0f;
            BF16x16 pk;
#pragma unroll
            for (int e = 0; e < 16; ++e) {
                const float p = __expf(se[e] - mnew);
                rs += p;
                pk.u[e] = f2bf(p);
            }
            rs += __shfl_xor(rs, 16, 32);
            const float aRow = __expf(mrow[qt] - mnew);
            lrow[qt] = lrow[qt] * aRow + rs;
            mrow[qt] = mnew;
            Pa[qt] = pk.v;

            // ---- rescale O (broadcast row alpha into C/D layout)
#pragma unroll
            for (int r = 0; r < 8; ++r) {
                const float al = __shfl(aRow, r + 8 * half, 32);
                o[qt][0][r] *= al; o[qt][1][r] *= al;
                o[qt][2][r] *= al; o[qt][3][r] *= al;
            }
        }

        // ---- V as B-operand (shared across subtiles)
        v16bf Vb[4];
        {
#pragma unroll
            for (int g = 0; g < 4; ++g) {
                BF16x16 tv;
#pragma unroll
                for (int e = 0; e < 16; ++e) {
                    const int key = 16 * (e >> 3) + 8 * half + (e & 7);
                    tv.u[e] = vs[key * DH + 16 * g + ln];
                }
                Vb[g] = tv.v;
            }
        }

        // ---- O += P(16x32) x V(32x16) per d-group, per active subtile
#pragma unroll
        for (int qt = 0; qt < 2; ++qt) {
            if (!act[qt]) continue;
#pragma unroll
            for (int g = 0; g < 4; ++g)
                o[qt][g] = __builtin_amdgcn_wmma_f32_16x16x32_bf16(false, Pa[qt], false, Vb[g],
                                                                   (short)0, o[qt][g], false, false);
        }
    }

    // ---- normalize and store (broadcast 1/l into C/D layout)
#pragma unroll
    for (int qt = 0; qt < 2; ++qt) {
        const float linv = 1.0f / lrow[qt];
#pragma unroll
        for (int r = 0; r < 8; ++r) {
            const float nv = __shfl(linv, r + 8 * half, 32);
            float* orow = O + baseF + (size_t)(q0 + 16 * qt + r + 8 * half) * DH;
            orow[ln]      = o[qt][0][r] * nv;
            orow[16 + ln] = o[qt][1][r] * nv;
            orow[32 + ln] = o[qt][2][r] * nv;
            orow[48 + ln] = o[qt][3][r] * nv;
        }
    }
}

extern "C" void kernel_launch(void* const* d_in, const int* in_sizes, int n_in,
                              void* d_out, int out_size, void* d_ws, size_t ws_size,
                              hipStream_t stream) {
    (void)in_sizes; (void)n_in; (void)out_size; (void)ws_size;
    const float* Q = (const float*)d_in[0];
    const float* K = (const float*)d_in[1];
    const float* V = (const float*)d_in[2];
    float* O = (float*)d_out;

    // Scratch layout (needs 16 MB): K^T bf16 [bh][d][s], V bf16 [bh][s][d]
    unsigned short* Kt = (unsigned short*)d_ws;
    unsigned short* Vb = Kt + (size_t)BH * SEQ * DH;

    const int nElem = BH * SEQ * DH;                    // 4,194,304
    cvt_bf16_kernel<<<dim3(nElem / 1024), 256, 0, stream>>>(K, V, Kt, Vb);
    sdpa_fa_wmma_bf16<<<dim3(SEQ / QBLK, BH), 256, 0, stream>>>(Q, Kt, Vb, O);
}
